// NystromAttention_54528904790348
// MI455X (gfx1250) — matrix operand order repaired and checked by reading, not compiled, not run on hardware
//
#include <hip/hip_runtime.h>
#include <math.h>

// ---------------------------------------------------------------------------
// Types for CDNA5 WMMA
// ---------------------------------------------------------------------------
typedef __bf16 bf16;
typedef __attribute__((ext_vector_type(16))) __bf16 v16bf;
typedef __attribute__((ext_vector_type(8)))  __bf16 v8bf;
typedef __attribute__((ext_vector_type(4)))  __bf16 v4bf;
typedef __attribute__((ext_vector_type(8)))  float  v8f;
typedef __attribute__((ext_vector_type(4)))  float  v4f;
typedef __attribute__((ext_vector_type(2)))  float  v2f;

// Block tiling: 256 threads = 8 wave32, wave grid 4x2, each wave 32x32 (2x2 WMMA tiles)
#define BM 128
#define BN 64
#define BK 32
#define LDAS 48   // LDS row stride (bf16 elems): 96 B -> 16B-aligned frag chunks, bank-spread
#define LDBS 48
#define LDASF 36  // LDS row stride (f32 elems): 144 B -> 16B-aligned b128 stores, 8B v2f frags
#define LDBSF 36

// ---------------------------------------------------------------------------
// Generic batched-strided GEMM, bf16 WMMA (fp32 in global, bf16 in LDS, fp32 acc)
// A is K-contiguous: A(m,k) = A[m*sa_m + k]            (true for all calls here)
// B_KC=true : B(k,n) = B[n*sb_n + k]  (K-contiguous)   -> vector loads along K
// B_KC=false: B(k,n) = B[k*sb_k + n]  (N-contiguous)   -> vector loads along N
//   mode 0: C = alpha * acc
//   mode 1: C = diag*I - acc
//   mode 2: C = acc + bias[n]
// Batch: z = blockIdx.z; b = z/bdiv; h = z%bdiv; per-operand (b,h) strides.
// Requires M%128==0, N%64==0, K%32==0 (true for every GEMM in this pipeline).
// ---------------------------------------------------------------------------
template <bool B_KC>
__global__ __launch_bounds__(256) void gemm_bf16(
    const float* __restrict__ A, const float* __restrict__ B, float* __restrict__ C,
    int K,
    long long sa_m, long long sb_k, long long sb_n, long long sc_m,
    int bdiv,
    long long sAb, long long sAh, long long sBb, long long sBh,
    long long sCb, long long sCh,
    float alpha, int mode, float diag, const float* __restrict__ bias)
{
  __shared__ bf16 As[BM * LDAS];
  __shared__ bf16 Bs[BN * LDBS];
  const int tid  = threadIdx.x;
  const int z    = blockIdx.z;
  const int bb   = z / bdiv, hh = z % bdiv;
  A += (long long)bb * sAb + (long long)hh * sAh;
  B += (long long)bb * sBb + (long long)hh * sBh;
  C += (long long)bb * sCb + (long long)hh * sCh;
  const int m0   = blockIdx.y * BM;
  const int n0   = blockIdx.x * BN;
  const int wave = tid >> 5, lane = tid & 31;
  const int wm   = wave & 3,  wn  = wave >> 2;     // 4 x 2 wave grid
  const int l16  = lane & 15, half = lane >> 4;

  v8f acc[2][2] = {};

  for (int k0 = 0; k0 < K; k0 += BK) {
    // ---- load phase: all global loads in flight before any LDS store ----
    v4f la[4];
#pragma unroll
    for (int j = 0; j < 4; ++j) {                 // A tile 128x32 = 1024 float4
      int idx = j * 256 + tid;
      int r = idx >> 3, c4 = (idx & 7) << 2;
      la[j] = *(const v4f*)&A[(long long)(m0 + r) * sa_m + (k0 + c4)];
    }
    v4f lb[2];
    if (B_KC) {
#pragma unroll
      for (int j = 0; j < 2; ++j) {               // B tile 64x32 = 512 float4 along K
        int idx = j * 256 + tid;
        int nn = idx >> 3, c4 = (idx & 7) << 2;
        lb[j] = *(const v4f*)&B[(long long)(n0 + nn) * sb_n + (k0 + c4)];
      }
    } else {
#pragma unroll
      for (int j = 0; j < 2; ++j) {               // 512 float4 along N
        int idx = j * 256 + tid;
        int kk = idx >> 4, n4 = (idx & 15) << 2;
        lb[j] = *(const v4f*)&B[(long long)(k0 + kk) * sb_k + (n0 + n4)];
      }
    }
    if (k0 + BK < K) { // hint next A tile toward cache (global_prefetch_b8)
      __builtin_prefetch(&A[(long long)(m0 + (tid >> 1)) * sa_m + (k0 + BK)], 0, 1);
    }
    __syncthreads();   // previous iteration's fragment reads done before overwrite

    // ---- store phase: convert fp32 -> bf16, packed LDS stores ----
#pragma unroll
    for (int j = 0; j < 4; ++j) {
      int idx = j * 256 + tid;
      int r = idx >> 3, c4 = (idx & 7) << 2;
      v4bf p;
#pragma unroll
      for (int e = 0; e < 4; ++e) p[e] = (bf16)la[j][e];
      *(v4bf*)&As[r * LDAS + c4] = p;             // ds_store_b64
    }
    if (B_KC) {
#pragma unroll
      for (int j = 0; j < 2; ++j) {
        int idx = j * 256 + tid;
        int nn = idx >> 3, c4 = (idx & 7) << 2;
        v4bf p;
#pragma unroll
        for (int e = 0; e < 4; ++e) p[e] = (bf16)lb[j][e];
        *(v4bf*)&Bs[nn * LDBS + c4] = p;          // ds_store_b64, Bs[n][k]
      }
    } else {
#pragma unroll
      for (int j = 0; j < 2; ++j) {
        int idx = j * 256 + tid;
        int kk = idx >> 4, n4 = (idx & 15) << 2;
#pragma unroll
        for (int e = 0; e < 4; ++e)               // transpose scatter
          Bs[(n4 + e) * LDBS + kk] = (bf16)lb[j][e];
      }
    }
    __syncthreads();

    // ---- fragments (ISA 7.12.2 layouts), two ds_load_b128 each ----
    v16bf afr[2], bfr[2];
#pragma unroll
    for (int t = 0; t < 2; ++t) {
      // A 16x32: element e -> K = (e<8?e:e+8) + 8*half
      const bf16* pa = &As[(wm * 32 + t * 16 + l16) * LDAS];
      v8bf a0 = *(const v8bf*)(pa + 8 * half);
      v8bf a1 = *(const v8bf*)(pa + 16 + 8 * half);
#pragma unroll
      for (int e = 0; e < 8; ++e) { afr[t][e] = a0[e]; afr[t][e + 8] = a1[e]; }
      // B 32x16: element e -> K = (e<8?e:e+8) + 8*half
      const bf16* pb = &Bs[(wn * 32 + t * 16 + l16) * LDBS];
      v8bf b0 = *(const v8bf*)(pb + 8 * half);
      v8bf b1 = *(const v8bf*)(pb + 16 + 8 * half);
#pragma unroll
      for (int e = 0; e < 8; ++e) { bfr[t][e] = b0[e]; bfr[t][e + 8] = b1[e]; }
    }
#pragma unroll
    for (int tm = 0; tm < 2; ++tm)
#pragma unroll
      for (int tn = 0; tn < 2; ++tn)
        acc[tm][tn] = __builtin_amdgcn_wmma_f32_16x16x32_bf16(
            false, afr[tm], false, bfr[tn], (short)0, acc[tm][tn], false, false);
  }

  // Epilogue. C/D layout: row = r + 8*half, col = l16.
#pragma unroll
  for (int tm = 0; tm < 2; ++tm)
#pragma unroll
    for (int tn = 0; tn < 2; ++tn)
#pragma unroll
      for (int r = 0; r < 8; ++r) {
        int gm = m0 + wm * 32 + tm * 16 + 8 * half + r;
        int gn = n0 + wn * 32 + tn * 16 + l16;
        float v = acc[tm][tn][r];
        if (mode == 0)      v *= alpha;
        else if (mode == 1) v = ((gm == gn) ? diag : 0.0f) - v;
        else                v += bias[gn];
        C[(long long)gm * sc_m + gn] = v;
      }
}

// ---------------------------------------------------------------------------
// Full-fp32 WMMA twin (v_wmma_f32_16x16x4_f32) for the Newton-Schulz pinv.
// A K-contiguous; B always N-contiguous here: B(k,n) = B[k*sb_k + n].
// ---------------------------------------------------------------------------
__global__ __launch_bounds__(256) void gemm_f32(
    const float* __restrict__ A, const float* __restrict__ B, float* __restrict__ C,
    int K,
    long long sa_m, long long sb_k, long long sc_m,
    int bdiv,
    long long sAb, long long sAh, long long sBb, long long sBh,
    long long sCb, long long sCh,
    float alpha, int mode, float diag, const float* __restrict__ bias)
{
  __shared__ float As[BM * LDASF];
  __shared__ float Bs[BN * LDBSF];
  const int tid  = threadIdx.x;
  const int z    = blockIdx.z;
  const int bb   = z / bdiv, hh = z % bdiv;
  A += (long long)bb * sAb + (long long)hh * sAh;
  B += (long long)bb * sBb + (long long)hh * sBh;
  C += (long long)bb * sCb + (long long)hh * sCh;
  const int m0   = blockIdx.y * BM;
  const int n0   = blockIdx.x * BN;
  const int wave = tid >> 5, lane = tid & 31;
  const int wm   = wave & 3,  wn  = wave >> 2;
  const int l16  = lane & 15, half = lane >> 4;

  v8f acc[2][2] = {};

  for (int k0 = 0; k0 < K; k0 += BK) {
    v4f la[4];
#pragma unroll
    for (int j = 0; j < 4; ++j) {
      int idx = j * 256 + tid;
      int r = idx >> 3, c4 = (idx & 7) << 2;
      la[j] = *(const v4f*)&A[(long long)(m0 + r) * sa_m + (k0 + c4)];
    }
    v4f lb[2];
#pragma unroll
    for (int j = 0; j < 2; ++j) {
      int idx = j * 256 + tid;
      int kk = idx >> 4, n4 = (idx & 15) << 2;
      lb[j] = *(const v4f*)&B[(long long)(k0 + kk) * sb_k + (n0 + n4)];
    }
    __syncthreads();
#pragma unroll
    for (int j = 0; j < 4; ++j) {
      int idx = j * 256 + tid;
      int r = idx >> 3, c4 = (idx & 7) << 2;
      *(v4f*)&As[r * LDASF + c4] = la[j];         // ds_store_b128
    }
#pragma unroll
    for (int j = 0; j < 2; ++j) {
      int idx = j * 256 + tid;
      int kk = idx >> 4, n4 = (idx & 15) << 2;
#pragma unroll
      for (int e = 0; e < 4; ++e)
        Bs[(n4 + e) * LDBSF + kk] = lb[j][e];     // transpose scatter
    }
    __syncthreads();

    // 8 K-sub-steps of 4; f32 A/B fragment: element e -> K = e + 2*half (+4*kk)
#pragma unroll
    for (int kk = 0; kk < 8; ++kk) {
      v2f a2[2], b2[2];
#pragma unroll
      for (int t = 0; t < 2; ++t) {
        a2[t] = *(const v2f*)&As[(wm * 32 + t * 16 + l16) * LDASF + kk * 4 + 2 * half];
        b2[t] = *(const v2f*)&Bs[(wn * 32 + t * 16 + l16) * LDBSF + kk * 4 + 2 * half];
      }
#pragma unroll
      for (int tm = 0; tm < 2; ++tm)
#pragma unroll
        for (int tn = 0; tn < 2; ++tn)
          acc[tm][tn] = __builtin_amdgcn_wmma_f32_16x16x4_f32(
              false, a2[tm], false, b2[tn], (short)0, acc[tm][tn], false, false);
    }
  }

#pragma unroll
  for (int tm = 0; tm < 2; ++tm)
#pragma unroll
    for (int tn = 0; tn < 2; ++tn)
#pragma unroll
      for (int r = 0; r < 8; ++r) {
        int gm = m0 + wm * 32 + tm * 16 + 8 * half + r;
        int gn = n0 + wn * 32 + tn * 16 + l16;
        float v = acc[tm][tn][r];
        if (mode == 0)      v *= alpha;
        else if (mode == 1) v = ((gm == gn) ? diag : 0.0f) - v;
        else                v += bias[gn];
        C[(long long)gm * sc_m + gn] = v;
      }
}

// ---------------------------------------------------------------------------
// Row softmax, one block (256 thr) per row.
// ---------------------------------------------------------------------------
__global__ __launch_bounds__(256) void row_softmax(float* __restrict__ d, int cols)
{
  __shared__ float red[256];
  const int tid = threadIdx.x;
  float* p = d + (long long)blockIdx.x * cols;
  float mx = -3.4e38f;
  for (int c = tid; c < cols; c += 256) mx = fmaxf(mx, p[c]);
  red[tid] = mx; __syncthreads();
  for (int s = 128; s > 0; s >>= 1) { if (tid < s) red[tid] = fmaxf(red[tid], red[tid + s]); __syncthreads(); }
  mx = red[0]; __syncthreads();
  float sum = 0.f;
  for (int c = tid; c < cols; c += 256) { float e = __expf(p[c] - mx); p[c] = e; sum += e; }
  red[tid] = sum; __syncthreads();
  for (int s = 128; s > 0; s >>= 1) { if (tid < s) red[tid] += red[tid + s]; __syncthreads(); }
  float inv = 1.f / red[0];
  for (int c = tid; c < cols; c += 256) p[c] *= inv;
}

// ---------------------------------------------------------------------------
// Landmark pooling: qL includes q-scale (d^-0.5 = 0.125) folded with mean 1/32.
// qkv layout [b, n, 1536]; q cols 0..511, k cols 512..1023.
// ---------------------------------------------------------------------------
__global__ __launch_bounds__(256) void landmarks(
    const float* __restrict__ qkv, float* __restrict__ qL, float* __restrict__ kL)
{
  int idx = blockIdx.x * 256 + threadIdx.x;  // 4*8*256*64 = 2^19
  int dd = idx & 63, mm = (idx >> 6) & 255, h = (idx >> 14) & 7, b = idx >> 17;
  const float* base = qkv + ((long long)b * 8192 + mm * 32) * 1536 + h * 64 + dd;
  float sq = 0.f, sk = 0.f;
#pragma unroll 4
  for (int j = 0; j < 32; ++j) { sq += base[(long long)j * 1536]; sk += base[(long long)j * 1536 + 512]; }
  qL[idx] = sq * (0.125f / 32.f);
  kL[idx] = sk * (1.f / 32.f);
}

// ---------------------------------------------------------------------------
// Pinv setup: global max of per-row / per-col abs sums of attn2, then z0 = x^T/(cm*rm)
// ---------------------------------------------------------------------------
__global__ void init2(float* p) { if (threadIdx.x < 2) p[threadIdx.x] = 0.f; }

__global__ __launch_bounds__(256) void pinv_norm(const float* __restrict__ x, float* __restrict__ mx)
{
  __shared__ float red[256];
  int id = blockIdx.x;                 // [0, 2*8192)
  int isCol = (id >= 8192) ? 1 : 0;
  int r = id & 8191;
  int bh = r >> 8, i = r & 255;
  const float* mb = x + ((long long)bh << 16);
  int j = threadIdx.x;
  float v = isCol ? fabsf(mb[(long long)j * 256 + i]) : fabsf(mb[(long long)i * 256 + j]);
  red[threadIdx.x] = v; __syncthreads();
  for (int s = 128; s > 0; s >>= 1) { if (threadIdx.x < s) red[threadIdx.x] += red[threadIdx.x + s]; __syncthreads(); }
  if (threadIdx.x == 0)
    atomicMax((unsigned int*)&mx[isCol], __float_as_uint(red[0]));  // sums are >= 0
}

__global__ __launch_bounds__(256) void pinv_init_z(
    const float* __restrict__ x, float* __restrict__ z, const float* __restrict__ mx)
{
  float inv = 1.f / (mx[0] * mx[1] + 1e-12f);
  long long idx = (long long)blockIdx.x * 256 + threadIdx.x;   // 32*256*256
  long long bh = idx >> 16;
  int i = (int)((idx >> 8) & 255), j = (int)(idx & 255);
  z[(bh << 16) + (long long)i * 256 + j] = x[(bh << 16) + (long long)j * 256 + i] * inv;
}

__global__ __launch_bounds__(256) void diag_minus(
    const float* __restrict__ in, float* __restrict__ out, float diag)
{
  long long idx = (long long)blockIdx.x * 256 + threadIdx.x;
  int i = (int)((idx >> 8) & 255), j = (int)(idx & 255);
  float v = -in[idx];
  if (i == j) v += diag;
  out[idx] = v;
}

// ---------------------------------------------------------------------------
// Depthwise conv residual over v (taps along n, per head) + head-merge transpose:
// comb[b, i, h*64+dd] = outh[b,h,i,dd] + sum_t w[h,t]*v[b, i+t-16, h, dd]
// ---------------------------------------------------------------------------
__global__ __launch_bounds__(256) void conv_combine(
    const float* __restrict__ qkv, const float* __restrict__ outh,
    const float* __restrict__ convw, float* __restrict__ comb)
{
  long long idx = (long long)blockIdx.x * 256 + threadIdx.x;   // 4*8*8192*64
  int dd = (int)(idx & 63);
  int i  = (int)((idx >> 6) & 8191);
  int h  = (int)((idx >> 19) & 7);
  int b  = (int)(idx >> 22);
  const float* w = convw + h * 33;
  float acc = outh[((((long long)(b * 8 + h)) * 8192 + i) << 6) + dd];
  const float* vcol = qkv + (long long)b * 8192 * 1536 + 1024 + h * 64 + dd;
#pragma unroll
  for (int t = 0; t < 33; ++t) {
    int ii = i + t - 16;
    if (ii >= 0 && ii < 8192) acc += w[t] * vcol[(long long)ii * 1536];
  }
  comb[((long long)b * 8192 + i) * 512 + h * 64 + dd] = acc;
}

// ---------------------------------------------------------------------------
// Host orchestration. Workspace layout (fp32 elements); needs ~931 MB.
// ---------------------------------------------------------------------------
extern "C" void kernel_launch(void* const* d_in, const int* in_sizes, int n_in,
                              void* d_out, int out_size, void* d_ws, size_t ws_size,
                              hipStream_t stream)
{
  (void)in_sizes; (void)n_in; (void)out_size; (void)ws_size;
  const float* x     = (const float*)d_in[0];  // [4,8192,512]
  const float* Wqkv  = (const float*)d_in[1];  // [512,1536]
  const float* Wout  = (const float*)d_in[2];  // [512,512]
  const float* bout  = (const float*)d_in[3];  // [512]
  const float* convw = (const float*)d_in[4];  // [8,1,33,1]
  float* out = (float*)d_out;                  // [4,8192,512]
  float* ws  = (float*)d_ws;

  float* qkv  = ws;                       // 50331648  [4,8192,1536]
  float* qL   = qkv  + 50331648LL;        // 524288    [4,8,256,64] (scaled)
  float* kL   = qL   + 524288LL;          // 524288
  float* s1   = kL   + 524288LL;          // 67108864  [4,8,8192,256]
  float* s2   = s1   + 67108864LL;        // 2097152   [4,8,256,256]
  float* s3   = s2   + 2097152LL;         // 67108864  [4,8,256,8192]
  float* z0   = s3   + 67108864LL;        // 2097152
  float* z1   = z0   + 2097152LL;         // 2097152
  float* xz   = z1   + 2097152LL;         // 2097152
  float* t1   = xz   + 2097152LL;         // 2097152
  float* t2   = t1   + 2097152LL;         // 2097152
  float* kv   = t2   + 2097152LL;         // 524288    [4,8,256,64]
  float* w2   = kv   + 524288LL;          // 524288    [4,8,256,64]
  float* outh = w2   + 524288LL;          // 16777216  [4,8,8192,64]
  float* comb = outh + 16777216LL;        // 16777216  [4,8192,512]
  float* mxs  = comb + 16777216LL;        // 2

  // 1) qkv = x @ Wqkv   (batch over b; Wqkv is N-contiguous)
  gemm_bf16<false><<<dim3(1536 / BN, 8192 / BM, 4), 256, 0, stream>>>(
      x, Wqkv, qkv, 512,
      512, 1536, 1, 1536,
      1, 8192LL * 512, 0, 0, 0, 8192LL * 1536, 0,
      1.f, 0, 0.f, nullptr);

  // 2) landmark pooling
  landmarks<<<2048, 256, 0, stream>>>(qkv, qL, kL);

  // 3) attn1 scores = scale * q @ kL^T    (kL^T is K-contiguous)
  gemm_bf16<true><<<dim3(256 / BN, 8192 / BM, 32), 256, 0, stream>>>(
      qkv, kL, s1, 64,
      1536, 1, 64, 256,
      8, 8192LL * 1536, 64, 8LL * 16384, 16384, 8LL * 8192 * 256, 8192LL * 256,
      0.125f, 0, 0.f, nullptr);
  row_softmax<<<262144, 256, 0, stream>>>(s1, 256);

  // 4) attn2 scores = qL @ kL^T
  gemm_bf16<true><<<dim3(256 / BN, 256 / BM, 32), 256, 0, stream>>>(
      qL, kL, s2, 64,
      64, 1, 64, 256,
      8, 8LL * 16384, 16384, 8LL * 16384, 16384, 8LL * 65536, 65536,
      1.f, 0, 0.f, nullptr);
  row_softmax<<<8192, 256, 0, stream>>>(s2, 256);

  // 5) attn3 scores = qL @ k^T   (k^T is K-contiguous, col stride 1536)
  gemm_bf16<true><<<dim3(8192 / BN, 256 / BM, 32), 256, 0, stream>>>(
      qL, qkv + 512, s3, 64,
      64, 1, 1536, 8192,
      8, 8LL * 16384, 16384, 8192LL * 1536, 64, 8LL * 256 * 8192, 256LL * 8192,
      1.f, 0, 0.f, nullptr);
  row_softmax<<<8192, 256, 0, stream>>>(s3, 8192);

  // 6) Moore-Penrose pinv of attn2 (fp32 WMMA path)
  init2<<<1, 32, 0, stream>>>(mxs);
  pinv_norm<<<16384, 256, 0, stream>>>(s2, mxs);
  pinv_init_z<<<8192, 256, 0, stream>>>(s2, z0, mxs);
  float* zc = z0; float* zn = z1;
  for (int it = 0; it < 6; ++it) {
    gemm_f32<<<dim3(256 / BN, 256 / BM, 32), 256, 0, stream>>>(
        s2, zc, xz, 256, 256, 256, 256,
        1, 65536, 0, 65536, 0, 65536, 0, 1.f, 0, 0.f, nullptr);
    diag_minus<<<8192, 256, 0, stream>>>(xz, t1, 7.f);
    gemm_f32<<<dim3(256 / BN, 256 / BM, 32), 256, 0, stream>>>(
        xz, t1, t2, 256, 256, 256, 256,
        1, 65536, 0, 65536, 0, 65536, 0, 1.f, 1, 15.f, nullptr);
    gemm_f32<<<dim3(256 / BN, 256 / BM, 32), 256, 0, stream>>>(
        xz, t2, t1, 256, 256, 256, 256,
        1, 65536, 0, 65536, 0, 65536, 0, 1.f, 1, 13.f, nullptr);
    gemm_f32<<<dim3(256 / BN, 256 / BM, 32), 256, 0, stream>>>(
        zc, t1, zn, 256, 256, 256, 256,
        1, 65536, 0, 65536, 0, 65536, 0, 0.25f, 0, 0.f, nullptr);
    float* tmp = zc; zc = zn; zn = tmp;
  }

  // 7) kv = attn3 @ v   (v is N-contiguous, row stride 1536)
  gemm_bf16<false><<<dim3(64 / BN, 256 / BM, 32), 256, 0, stream>>>(
      s3, qkv + 1024, kv, 8192,
      8192, 1536, 1, 64,
      8, 8LL * 256 * 8192, 256LL * 8192, 8192LL * 1536, 64, 8LL * 16384, 16384,
      1.f, 0, 0.f, nullptr);

  // 8) w2 = pinv(attn2) @ kv   (fp32 WMMA, small)
  gemm_f32<<<dim3(64 / BN, 256 / BM, 32), 256, 0, stream>>>(
      zc, kv, w2, 256, 256, 64, 64,
      1, 65536, 0, 16384, 0, 16384, 0,
      1.f, 0, 0.f, nullptr);

  // 9) outh = attn1 @ w2
  gemm_bf16<false><<<dim3(64 / BN, 8192 / BM, 32), 256, 0, stream>>>(
      s1, w2, outh, 256,
      256, 64, 1, 64,
      1, 8192LL * 256, 0, 16384, 0, 8192LL * 64, 0,
      1.f, 0, 0.f, nullptr);

  // 10) depthwise conv residual + merge heads
  conv_combine<<<65536, 256, 0, stream>>>(qkv, outh, convw, comb);

  // 11) out = comb @ Wout + bout   (flatten b,n -> M = 32768)
  gemm_bf16<false><<<dim3(512 / BN, 32768 / BM, 1), 256, 0, stream>>>(
      comb, Wout, out, 512,
      512, 512, 1, 512,
      1, 0, 0, 0, 0, 0, 0,
      1.f, 2, 0.f, bout);
}
